// Summarization_Model_72258529788594
// MI455X (gfx1250) — compile-verified, matrix operand order checked by reading, not run on hardware
//
#include <hip/hip_runtime.h>

// ---------------------------------------------------------------------------
// MI455X (gfx1250) implementation of the pointer-generator decoder.
// E=128, H=256, B=16, L=2048, T=100.
//
// Strategy (from MI455X roofline):
//  * enc_feat = es @ Wh^T precomputed ONCE with v_wmma_f32_16x16x32_bf16,
//    stored bf16 (16.8 MB).  encoder_states kept as a bf16 copy (33.5 MB).
//    Both stay resident in the 192 MB L2 across all 100 steps, and bf16
//    halves the dominant per-step streaming traffic.
//  * B=16 == WMMA M dimension: every dense matmul (x, LSTM gates, dec_feat,
//    out) is done with 16x16 WMMA tiles, bf16 operands / f32 accumulators.
//  * Recurrent part (one block, 8 waves): gate tiles are assigned so that
//    i/f/g/o for the same hidden unit live in the same lane's accumulators,
//    so the LSTM pointwise runs entirely in registers.
//  * The attn-weighted context reduction streams encoder_states tiles
//    through LDS with the gfx1250 async copy engine
//    (global_load_async_to_lds_b128 + s_wait_asynccnt), double-buffered so
//    tile i+1 is in flight while tile i is accumulated from LDS.
// ---------------------------------------------------------------------------

typedef __bf16 bf16;
typedef __attribute__((ext_vector_type(16))) __bf16        v16bf;
typedef __attribute__((ext_vector_type(8)))  float         v8f;
typedef __attribute__((ext_vector_type(4)))  unsigned int  u32x4;
typedef __attribute__((ext_vector_type(4)))  float         f32x4;

union Frag16 { u32x4 q[2]; v16bf v; };
union Pack8  { u32x4 q;    bf16 e[8]; };

// WMMA 16x16x32 bf16 A/B fragment loader.
// ISA layout (16-bit A 16x32): lane<16 holds line=lane&15, k = {0..7, 16..23};
// lane>=16 holds k = {8..15, 24..31}. Two contiguous 8-elem runs => 2x b128.
// 'line' = A row (row-major MxK) or B column via W row-major [N x K] (B=W^T).
__device__ __forceinline__ v16bf frag_bf16(const bf16* W, int ld, int line,
                                           int k0, int half) {
  const bf16* p = W + (size_t)line * ld + k0 + half * 8;
  Frag16 f;
  f.q[0] = *(const u32x4*)(p);
  f.q[1] = *(const u32x4*)(p + 16);
  return f.v;
}

// Same fragment but converting from an f32 source on the fly.
__device__ __forceinline__ v16bf frag_f32(const float* W, int ld, int line,
                                          int k0, int half) {
  const float* p = W + (size_t)line * ld + k0 + half * 8;
  f32x4 a0 = *(const f32x4*)(p);
  f32x4 a1 = *(const f32x4*)(p + 4);
  f32x4 b0 = *(const f32x4*)(p + 16);
  f32x4 b1 = *(const f32x4*)(p + 20);
  v16bf v;
#pragma unroll
  for (int i = 0; i < 4; i++) {
    v[i]      = (bf16)a0[i];
    v[4 + i]  = (bf16)a1[i];
    v[8 + i]  = (bf16)b0[i];
    v[12 + i] = (bf16)b1[i];
  }
  return v;
}

__device__ __forceinline__ v8f wmma_bf16(v16bf a, v16bf b, v8f c) {
  return __builtin_amdgcn_wmma_f32_16x16x32_bf16(false, a, false, b,
                                                 (short)0, c, false, false);
}

__device__ __forceinline__ float sigmoidf(float x) {
  return 1.0f / (1.0f + __expf(-x));
}

// gfx1250 async copy: LDS[lane's loff] <= 16 bytes at MEM[lane's gaddr].
// Generic shared pointers carry the LDS byte address in their low 32 bits
// tracked with ASYNCcnt.
__device__ __forceinline__ void async_load_b128_to_lds(const bf16* g,
                                                       const bf16* l) {
  unsigned loff = (unsigned)(size_t)l;
  asm volatile("global_load_async_to_lds_b128 %0, %1, off"
               :: "v"(loff), "v"(g) : "memory");
}
__device__ __forceinline__ void wait_async_le2() {
  asm volatile("s_wait_asynccnt 0x2" ::: "memory");
}
__device__ __forceinline__ void wait_async_le0() {
  asm volatile("s_wait_asynccnt 0x0" ::: "memory");
}

// ---------------------------------------------------------------------------
__global__ __launch_bounds__(256)
void f32_to_bf16_kernel(const float* __restrict__ in, bf16* __restrict__ out,
                        int n) {
  int i = blockIdx.x * 256 + threadIdx.x;
  if (i < n) out[i] = (bf16)in[i];
}

// enc_feat[b,l,h] = sum_e es[b,l,e] * Wh[h,e] + bh[h].
// GEMM M=32768, N=256, K=512. One wave = one 16x64 tile. grid (256, 4).
__global__ __launch_bounds__(256)
void enc_feat_kernel(const bf16* __restrict__ es_bf,   // [32768, 512]
                     const bf16* __restrict__ Wh_bf,   // [256, 512]
                     const float* __restrict__ bh,     // [256]
                     bf16* __restrict__ ef_bf) {       // [32768, 256]
  const int wave = threadIdx.x >> 5, lane = threadIdx.x & 31;
  const int half = lane >> 4, idx = lane & 15;
  const int mtile = blockIdx.x * 8 + wave;
  const int nbase = blockIdx.y * 64;
  const int mrow  = mtile * 16 + idx;

  v8f acc[4];
#pragma unroll
  for (int j = 0; j < 4; j++) {
    float b = bh[nbase + j * 16 + idx];
#pragma unroll
    for (int g = 0; g < 8; g++) acc[j][g] = b;
  }
  for (int kc = 0; kc < 16; kc++) {
    v16bf a = frag_bf16(es_bf, 512, mrow, kc * 32, half);
#pragma unroll
    for (int j = 0; j < 4; j++) {
      v16bf b = frag_bf16(Wh_bf, 512, nbase + j * 16 + idx, kc * 32, half);
      acc[j] = wmma_bf16(a, b, acc[j]);
    }
  }
#pragma unroll
  for (int j = 0; j < 4; j++) {
    int n = nbase + j * 16 + idx;
#pragma unroll
    for (int g = 0; g < 8; g++) {
      int m = mtile * 16 + g + half * 8;
      ef_bf[(size_t)m * 256 + n] = (bf16)acc[j][g];
    }
  }
}

// Recurrent part of one timestep (1 block, 8 waves):
//   x = inp_t @ Wx[:, :E]^T + bx                (ctx carried is always zero)
//   gates = x @ Wih^T + h @ Whh^T + bih + bhh;  LSTM pointwise
//   dec_feat = [h_new, c_new] @ Ws^T + bs
__global__ __launch_bounds__(256)
void step_recurrent_kernel(const float* __restrict__ inp_t,  // [16,128]
                           const bf16* __restrict__ Wx_bf,   // [128,640]
                           const float* __restrict__ bx,
                           const bf16* __restrict__ Wih_bf,  // [1024,128]
                           const float* __restrict__ bih,
                           const bf16* __restrict__ Whh_bf,  // [1024,256]
                           const float* __restrict__ bhh,
                           const bf16* __restrict__ Ws_bf,   // [256,512]
                           const float* __restrict__ bs,
                           float* __restrict__ ws_h,         // [16,256] in/out
                           float* __restrict__ ws_c,         // [16,256] in/out
                           float* __restrict__ ws_x,         // [16,128] out
                           float* __restrict__ ws_df) {      // [16,256] out
  __shared__ __align__(16) bf16 x_bf[16 * 128];
  __shared__ __align__(16) bf16 hprev_bf[16 * 256];
  __shared__ __align__(16) bf16 hc_bf[16 * 512];

  const int tid = threadIdx.x;
  const int wave = tid >> 5, lane = tid & 31;
  const int half = lane >> 4, idx = lane & 15;

  for (int i = tid; i < 16 * 256; i += 256) hprev_bf[i] = (bf16)ws_h[i];
  __syncthreads();

  // ---- x: M=16, N=128 (1 tile/wave), K=128 ----
  {
    int n0 = wave * 16;
    v8f acc;
    float b = bx[n0 + idx];
#pragma unroll
    for (int g = 0; g < 8; g++) acc[g] = b;
    for (int kc = 0; kc < 4; kc++) {
      v16bf a  = frag_f32(inp_t, 128, idx, kc * 32, half);
      v16bf bb = frag_bf16(Wx_bf, 640, n0 + idx, kc * 32, half);
      acc = wmma_bf16(a, bb, acc);
    }
    int n = n0 + idx;
#pragma unroll
    for (int g = 0; g < 8; g++) {
      int m = g + half * 8;
      ws_x[m * 128 + n] = acc[g];
      x_bf[m * 128 + n] = (bf16)acc[g];
    }
  }
  __syncthreads();

  // ---- gates: 8 tiles/wave arranged so i/f/g/o share a lane ----
  v8f gacc[2][4];
#pragma unroll
  for (int j = 0; j < 2; j++)
#pragma unroll
    for (int k = 0; k < 4; k++) {
      int n = wave * 32 + j * 16 + k * 256 + idx;
      float b = bih[n] + bhh[n];
#pragma unroll
      for (int g = 0; g < 8; g++) gacc[j][k][g] = b;
    }
  for (int kc = 0; kc < 4; kc++) {          // K=128 over x
    v16bf a = frag_bf16(x_bf, 128, idx, kc * 32, half);
#pragma unroll
    for (int j = 0; j < 2; j++)
#pragma unroll
      for (int k = 0; k < 4; k++) {
        v16bf bb = frag_bf16(Wih_bf, 128, wave * 32 + j * 16 + k * 256 + idx,
                             kc * 32, half);
        gacc[j][k] = wmma_bf16(a, bb, gacc[j][k]);
      }
  }
  for (int kc = 0; kc < 8; kc++) {          // K=256 over h_prev
    v16bf a = frag_bf16(hprev_bf, 256, idx, kc * 32, half);
#pragma unroll
    for (int j = 0; j < 2; j++)
#pragma unroll
      for (int k = 0; k < 4; k++) {
        v16bf bb = frag_bf16(Whh_bf, 256, wave * 32 + j * 16 + k * 256 + idx,
                             kc * 32, half);
        gacc[j][k] = wmma_bf16(a, bb, gacc[j][k]);
      }
  }

  // ---- LSTM pointwise in registers (torch gate order i,f,g,o) ----
#pragma unroll
  for (int j = 0; j < 2; j++) {
    int hh = wave * 32 + j * 16 + idx;
#pragma unroll
    for (int g = 0; g < 8; g++) {
      int m = g + half * 8;
      float ig = sigmoidf(gacc[j][0][g]);
      float fg = sigmoidf(gacc[j][1][g]);
      float gg = tanhf(gacc[j][2][g]);
      float og = sigmoidf(gacc[j][3][g]);
      float cn = fg * ws_c[m * 256 + hh] + ig * gg;
      float hn = og * tanhf(cn);
      ws_c[m * 256 + hh] = cn;
      ws_h[m * 256 + hh] = hn;
      hc_bf[m * 512 + hh]       = (bf16)hn;
      hc_bf[m * 512 + 256 + hh] = (bf16)cn;
    }
  }
  __syncthreads();

  // ---- dec_feat: M=16, N=256 (2 tiles/wave), K=512 ----
#pragma unroll
  for (int j = 0; j < 2; j++) {
    int n0 = wave * 32 + j * 16;
    v8f acc;
    float b = bs[n0 + idx];
#pragma unroll
    for (int g = 0; g < 8; g++) acc[g] = b;
    for (int kc = 0; kc < 16; kc++) {
      v16bf a  = frag_bf16(hc_bf, 512, idx, kc * 32, half);
      v16bf bb = frag_bf16(Ws_bf, 512, n0 + idx, kc * 32, half);
      acc = wmma_bf16(a, bb, acc);
    }
    int n = n0 + idx;
#pragma unroll
    for (int g = 0; g < 8; g++) ws_df[(g + half * 8) * 256 + n] = acc[g];
  }
}

// e[b,l] = sum_h v[h]*tanh(enc_feat[b,l,h] + dec_feat[b,h]).  grid (16, 8).
__global__ __launch_bounds__(256)
void attn_scores_kernel(const bf16* __restrict__ ef_bf,   // [16,2048,256]
                        const float* __restrict__ ws_df,  // [16,256]
                        const float* __restrict__ vvec,   // [256]
                        float* __restrict__ ws_e) {       // [16,2048]
  __shared__ float df_s[256];
  __shared__ float v_s[256];
  const int b = blockIdx.x;
  const int l = blockIdx.y * 256 + threadIdx.x;
  df_s[threadIdx.x] = ws_df[b * 256 + threadIdx.x];
  v_s[threadIdx.x]  = vvec[threadIdx.x];
  __syncthreads();

  const bf16* row = ef_bf + ((size_t)b * 2048 + l) * 256;
  float acc = 0.0f;
  for (int h0 = 0; h0 < 256; h0 += 8) {
    Pack8 u;
    u.q = *(const u32x4*)(row + h0);
#pragma unroll
    for (int i = 0; i < 8; i++)
      acc += v_s[h0 + i] * tanhf((float)u.e[i] + df_s[h0 + i]);
  }
  ws_e[b * 2048 + l] = acc;
}

// Softmax over L + mask renorm + attn output + ctx chunk.  grid (16, 4).
// ctx accumulation streams 32x128 bf16 tiles of encoder_states through LDS
// with the async copy engine, double-buffered.
__global__ __launch_bounds__(256)
void ctx_kernel(const float* __restrict__ ws_e,   // [16,2048]
                const float* __restrict__ mask,   // [16,2048]
                const bf16* __restrict__ es_bf,   // [16,2048,512]
                float* __restrict__ ws_ctx,       // [16,512]
                float* __restrict__ attn_out) {   // [16,2048] for this t
  __shared__ float w_s[2048];
  __shared__ float red[256];
  __shared__ __align__(16) bf16 tile[2][32 * 128];

  const int b = blockIdx.x, chunk = blockIdx.y;
  const int tid = threadIdx.x;
  const int c0 = chunk * 128;
  const float* e  = ws_e + b * 2048;
  const float* mk = mask + b * 2048;

  // per-thread async piece: rows r0 and r0+16, 8 contiguous cols at cseg.
  const int r0   = tid >> 4;
  const int cseg = (tid & 15) * 8;
  const bf16* g0 = es_bf + ((size_t)b * 2048 + r0) * 512 + c0 + cseg;
  const bf16* l0 = &tile[0][r0 * 128 + cseg];
  const bf16* l1 = &tile[1][r0 * 128 + cseg];

  // kick off tile 0 (overlaps with the softmax-stat phase below)
  async_load_b128_to_lds(g0, l0);
  async_load_b128_to_lds(g0 + 16 * 512, l0 + 16 * 128);

  // ---- softmax stats over L ----
  float mx = -1e30f;
  for (int l = tid; l < 2048; l += 256) mx = fmaxf(mx, e[l]);
  red[tid] = mx;
  __syncthreads();
  for (int s = 128; s > 0; s >>= 1) {
    if (tid < s) red[tid] = fmaxf(red[tid], red[tid + s]);
    __syncthreads();
  }
  mx = red[0];
  __syncthreads();

  float sm = 0.0f;
  for (int l = tid; l < 2048; l += 256) {
    float w = __expf(e[l] - mx) * mk[l];
    w_s[l] = w;
    sm += w;
  }
  red[tid] = sm;
  __syncthreads();
  for (int s = 128; s > 0; s >>= 1) {
    if (tid < s) red[tid] += red[tid + s];
    __syncthreads();
  }
  const float inv = 1.0f / red[0];
  __syncthreads();

  if (chunk == 0)
    for (int l = tid; l < 2048; l += 256) attn_out[b * 2048 + l] = w_s[l] * inv;

  // ---- ctx[b,c] = sum_l attn[b,l] * es[b,l,c], double-buffered tiles ----
  const int cc = tid & 127;
  const int p  = tid >> 7;
  float acc = 0.0f;
  for (int i = 0; i < 64; i++) {
    if (i + 1 < 64) {  // stream tile i+1 while tile i is consumed
      const bf16* g = g0 + (size_t)(i + 1) * 32 * 512;
      const bf16* l = (i & 1) ? l0 : l1;
      async_load_b128_to_lds(g, l);
      async_load_b128_to_lds(g + 16 * 512, l + 16 * 128);
      wait_async_le2();   // retire the two ops of tile i
    } else {
      wait_async_le0();
    }
    __syncthreads();
    const bf16* tb = tile[i & 1];
    const float* wrow = w_s + i * 32;
#pragma unroll
    for (int lr = p; lr < 32; lr += 2)
      acc += wrow[lr] * (float)tb[lr * 128 + cc];
    __syncthreads();
  }
  red[tid] = acc;
  __syncthreads();
  if (tid < 128)
    ws_ctx[b * 512 + c0 + tid] = (red[tid] + red[tid + 128]) * inv;
}

// out = [h_new, ctx] @ Wout^T + bout (WMMA) ; p_gen = sigmoid([ctx,h,c,x]·Wpg)
__global__ __launch_bounds__(256)
void out_pgen_kernel(const float* __restrict__ ws_h,
                     const float* __restrict__ ws_c,
                     const float* __restrict__ ws_ctx,
                     const float* __restrict__ ws_x,
                     const bf16* __restrict__ Wout_bf,  // [256,768]
                     const float* __restrict__ bout,
                     const float* __restrict__ Wpg,     // [1152]
                     const float* __restrict__ bpg,
                     float* __restrict__ out_t,         // [16,256]
                     float* __restrict__ pgen_t) {      // [16]
  __shared__ __align__(16) bf16 a_bf[16 * 768];
  const int tid = threadIdx.x;
  for (int i = tid; i < 16 * 256; i += 256) {
    int m = i >> 8, n = i & 255;
    a_bf[m * 768 + n] = (bf16)ws_h[i];
  }
  for (int i = tid; i < 16 * 512; i += 256) {
    int m = i >> 9, n = i & 511;
    a_bf[m * 768 + 256 + n] = (bf16)ws_ctx[i];
  }
  __syncthreads();

  const int wave = tid >> 5, lane = tid & 31;
  const int half = lane >> 4, idx = lane & 15;
#pragma unroll
  for (int j = 0; j < 2; j++) {
    int n0 = wave * 32 + j * 16;
    v8f acc;
    float b = bout[n0 + idx];
#pragma unroll
    for (int g = 0; g < 8; g++) acc[g] = b;
    for (int kc = 0; kc < 24; kc++) {
      v16bf a = frag_bf16(a_bf, 768, idx, kc * 32, half);
      v16bf w = frag_bf16(Wout_bf, 768, n0 + idx, kc * 32, half);
      acc = wmma_bf16(a, w, acc);
    }
    int n = n0 + idx;
#pragma unroll
    for (int g = 0; g < 8; g++) out_t[(g + half * 8) * 256 + n] = acc[g];
  }

  if (tid < 16) {  // p_gen: order [ctx, h_new, c_new, x] per reference
    int b = tid;
    float acc = bpg[0];
    for (int k = 0; k < 512; k++) acc += ws_ctx[b * 512 + k] * Wpg[k];
    for (int k = 0; k < 256; k++) acc += ws_h[b * 256 + k] * Wpg[512 + k];
    for (int k = 0; k < 256; k++) acc += ws_c[b * 256 + k] * Wpg[768 + k];
    for (int k = 0; k < 128; k++) acc += ws_x[b * 128 + k] * Wpg[1024 + k];
    pgen_t[b] = sigmoidf(acc);
  }
}

// ---------------------------------------------------------------------------
extern "C" void kernel_launch(void* const* d_in, const int* in_sizes, int n_in,
                              void* d_out, int out_size, void* d_ws,
                              size_t ws_size, hipStream_t stream) {
  (void)in_sizes; (void)n_in; (void)out_size; (void)ws_size;
  const float* dec_in     = (const float*)d_in[0];
  const float* h0         = (const float*)d_in[1];
  const float* c0         = (const float*)d_in[2];
  const float* enc_states = (const float*)d_in[3];
  const float* mask       = (const float*)d_in[4];
  const float* Wh  = (const float*)d_in[5];
  const float* bh  = (const float*)d_in[6];
  const float* Ws  = (const float*)d_in[7];
  const float* bs  = (const float*)d_in[8];
  const float* v   = (const float*)d_in[9];
  const float* Wx  = (const float*)d_in[10];
  const float* bx  = (const float*)d_in[11];
  const float* Wih = (const float*)d_in[12];
  const float* bih = (const float*)d_in[13];
  const float* Whh = (const float*)d_in[14];
  const float* bhh = (const float*)d_in[15];
  const float* Wpg = (const float*)d_in[16];
  const float* bpg = (const float*)d_in[17];
  const float* Wout = (const float*)d_in[18];
  const float* bout = (const float*)d_in[19];

  // d_out: outputs[100,16,256] | h_f[16,256] | c_f[16,256] |
  //        attn[100,16,2048] | p_gens[100,16,1]
  float* out_outputs = (float*)d_out;
  float* out_hf   = out_outputs + 100 * 16 * 256;
  float* out_cf   = out_hf + 16 * 256;
  float* out_attn = out_cf + 16 * 256;
  float* out_pgen = out_attn + (size_t)100 * 16 * 2048;

  // workspace carve-out (~52 MB)
  char* ws = (char*)d_ws;
  size_t off = 0;
  auto alloc = [&](size_t bytes) -> char* {
    char* p = ws + off;
    off += (bytes + 255) & ~(size_t)255;
    return p;
  };
  bf16*  es_bf   = (bf16*)alloc((size_t)16 * 2048 * 512 * 2);
  bf16*  ef_bf   = (bf16*)alloc((size_t)16 * 2048 * 256 * 2);
  float* ws_e    = (float*)alloc((size_t)16 * 2048 * 4);
  float* ws_h    = (float*)alloc(16 * 256 * 4);
  float* ws_c    = (float*)alloc(16 * 256 * 4);
  float* ws_x    = (float*)alloc(16 * 128 * 4);
  float* ws_df   = (float*)alloc(16 * 256 * 4);
  float* ws_ctx  = (float*)alloc(16 * 512 * 4);
  bf16*  Wx_bf   = (bf16*)alloc(128 * 640 * 2);
  bf16*  Wih_bf  = (bf16*)alloc(1024 * 128 * 2);
  bf16*  Whh_bf  = (bf16*)alloc(1024 * 256 * 2);
  bf16*  Ws_bf   = (bf16*)alloc(256 * 512 * 2);
  bf16*  Wh_bf   = (bf16*)alloc(256 * 512 * 2);
  bf16*  Wout_bf = (bf16*)alloc(256 * 768 * 2);

  auto conv = [&](const float* in, bf16* outp, int n) {
    f32_to_bf16_kernel<<<(n + 255) / 256, 256, 0, stream>>>(in, outp, n);
  };
  conv(enc_states, es_bf, 16 * 2048 * 512);
  conv(Wx, Wx_bf, 128 * 640);
  conv(Wih, Wih_bf, 1024 * 128);
  conv(Whh, Whh_bf, 1024 * 256);
  conv(Ws, Ws_bf, 256 * 512);
  conv(Wh, Wh_bf, 256 * 512);
  conv(Wout, Wout_bf, 256 * 768);

  hipMemcpyAsync(ws_h, h0, 16 * 256 * 4, hipMemcpyDeviceToDevice, stream);
  hipMemcpyAsync(ws_c, c0, 16 * 256 * 4, hipMemcpyDeviceToDevice, stream);

  enc_feat_kernel<<<dim3(256, 4), 256, 0, stream>>>(es_bf, Wh_bf, bh, ef_bf);

  for (int t = 0; t < 100; t++) {
    step_recurrent_kernel<<<1, 256, 0, stream>>>(
        dec_in + (size_t)t * 16 * 128, Wx_bf, bx, Wih_bf, bih, Whh_bf, bhh,
        Ws_bf, bs, ws_h, ws_c, ws_x, ws_df);
    attn_scores_kernel<<<dim3(16, 8), 256, 0, stream>>>(ef_bf, ws_df, v, ws_e);
    ctx_kernel<<<dim3(16, 4), 256, 0, stream>>>(
        ws_e, mask, es_bf, ws_ctx, out_attn + (size_t)t * 16 * 2048);
    out_pgen_kernel<<<1, 256, 0, stream>>>(
        ws_h, ws_c, ws_ctx, ws_x, Wout_bf, bout, Wpg, bpg,
        out_outputs + (size_t)t * 16 * 256, out_pgen + t * 16);
  }

  hipMemcpyAsync(out_hf, ws_h, 16 * 256 * 4, hipMemcpyDeviceToDevice, stream);
  hipMemcpyAsync(out_cf, ws_c, 16 * 256 * 4, hipMemcpyDeviceToDevice, stream);
}
